// CausalMatchingModule_343597384441
// MI455X (gfx1250) — matrix operand order repaired and checked by reading, not compiled, not run on hardware
//
#include <hip/hip_runtime.h>
#include <hip/hip_bf16.h>
#include <math.h>

// Problem constants (from reference)
#define B_  64
#define N_  16
#define D_  512
#define P_  64
#define L_  8
#define D2_ 256
#define K1_ 1024          // 2*D
#define ROWS_TOTAL (B_*P_*L_)   // 32768
#define TILES (ROWS_TOTAL/16)   // 2048

typedef __attribute__((ext_vector_type(16))) _Float16 v16h;
typedef __attribute__((ext_vector_type(8)))  _Float16 v8h;
typedef __attribute__((ext_vector_type(8)))  float    v8f;

// ---------------- prep kernels ----------------

__global__ void kCvtF16(const float* __restrict__ src, _Float16* __restrict__ dst, int n) {
    int i = blockIdx.x * blockDim.x + threadIdx.x;
    int stride = gridDim.x * blockDim.x;
    for (; i < n; i += stride) dst[i] = (_Float16)src[i];
}

// src (rows, cols) row-major f32 -> dst (cols, rows) row-major f16
__global__ void kTransF16(const float* __restrict__ src, _Float16* __restrict__ dst,
                          int rows, int cols) {
    int n = rows * cols;
    int i = blockIdx.x * blockDim.x + threadIdx.x;
    int stride = gridDim.x * blockDim.x;
    for (; i < n; i += stride) {
        int r = i / cols, c = i - r * cols;
        dst[c * rows + r] = (_Float16)src[i];
    }
}

// ---------------- main WMMA MLP kernel ----------------
// grid.x = 2 * TILES; block = 256 threads (8 wave32)
// For each 16-row tile: gather pair tile -> LDS, layer1 WMMA (K=1024,N=512),
// layer2 WMMA (K=512,N=256), layer3 dot+sigmoid -> sOut.
__global__ void __launch_bounds__(256)
kPathMLP(const _Float16* __restrict__ imgH,
         const _Float16* __restrict__ textH,
         const int*      __restrict__ paths,   // (P, L+1) int32
         const _Float16* __restrict__ W1t,     // (512, 1024) = W1^T
         const float*    __restrict__ b1,      // (512)
         const _Float16* __restrict__ W2t,     // (256, 512)  = W2^T
         const float*    __restrict__ b2,      // (256)
         const float*    __restrict__ W3,      // (256)
         const float*    __restrict__ b3,      // (1)
         float*          __restrict__ sOut)    // (2, 32768)
{
    __shared__ _Float16 sA [16 * K1_];   // 32 KB pair tile, row-major (16 x 1024)
    __shared__ _Float16 sH1[16 * D_];    // 16 KB layer1 out (16 x 512)
    __shared__ _Float16 sH2[16 * D2_];   //  8 KB layer2 out (16 x 256)
    __shared__ int sRow[32];             // per-row src/dst feature offsets

    const int tid  = threadIdx.x;
    const int feat = blockIdx.x >> 11;           // 0=img, 1=text
    const int tile = blockIdx.x & (TILES - 1);
    const _Float16* featsH = feat ? textH : imgH;

    // --- step 0: row -> (b, src, dst) offsets
    if (tid < 16) {
        int r   = tile * 16 + tid;
        int b   = r >> 9;          // / (P*L)
        int rem = r & 511;
        int p   = rem >> 3;
        int l   = rem & 7;
        int src = paths[p * (L_ + 1) + l];
        int dst = paths[p * (L_ + 1) + l + 1];
        sRow[tid * 2 + 0] = (b * N_ + src) * D_;
        sRow[tid * 2 + 1] = (b * N_ + dst) * D_;
    }
    __syncthreads();

    // --- step 1: gather 16 x 1024 pair tile into LDS (16B chunks)
    for (int q = tid; q < 16 * 128; q += 256) {
        int m = q >> 7;            // row in tile
        int c = q & 127;           // 8-f16 chunk within 1024
        int off = sRow[m * 2 + (c >> 6)] + (c & 63) * 8;
        *(uint4*)&sA[m * K1_ + c * 8] = *(const uint4*)&featsH[off];
    }
    __syncthreads();

    const int wave = tid >> 5;
    const int lane = tid & 31;
    const int half = lane >> 4;
    const int ln   = lane & 15;

    // --- step 2: layer1, 16x1024 @ 1024x512 via v_wmma_f32_16x16x32_f16
    for (int nt = wave; nt < 32; nt += 8) {
        v8f acc = {};
        const int n = nt * 16 + ln;
        for (int kt = 0; kt < 32; ++kt) {
            const int k0 = kt * 32;
            // A fragment: lane holds row ln, K = k0 + half*16 .. +15 (contiguous)
            v8h alo = *(const v8h*)&sA[ln * K1_ + k0 + half * 16];
            v8h ahi = *(const v8h*)&sA[ln * K1_ + k0 + half * 16 + 8];
            v16h afrag;
#pragma unroll
            for (int i = 0; i < 8; ++i) { afrag[i] = alo[i]; afrag[i + 8] = ahi[i]; }
            // B fragment: lane holds col n, K = k0 + half*16 .. +15 (contiguous in W1t)
            v16h bfrag = *(const v16h*)&W1t[n * K1_ + k0 + half * 16];
            acc = __builtin_amdgcn_wmma_f32_16x16x32_f16(
                    false, afrag, false, bfrag, (short)0, acc, false, false);
        }
        const float bias = b1[n];
#pragma unroll
        for (int v = 0; v < 8; ++v) {
            int row = v + half * 8;
            float x = acc[v] + bias;
            sH1[row * D_ + n] = (_Float16)fmaxf(x, 0.0f);
        }
    }
    __syncthreads();

    // --- step 3: layer2, 16x512 @ 512x256
    for (int nt = wave; nt < 16; nt += 8) {
        v8f acc = {};
        const int n = nt * 16 + ln;
        for (int kt = 0; kt < 16; ++kt) {
            const int k0 = kt * 32;
            v8h alo = *(const v8h*)&sH1[ln * D_ + k0 + half * 16];
            v8h ahi = *(const v8h*)&sH1[ln * D_ + k0 + half * 16 + 8];
            v16h afrag;
#pragma unroll
            for (int i = 0; i < 8; ++i) { afrag[i] = alo[i]; afrag[i + 8] = ahi[i]; }
            v16h bfrag = *(const v16h*)&W2t[n * D_ + k0 + half * 16];
            acc = __builtin_amdgcn_wmma_f32_16x16x32_f16(
                    false, afrag, false, bfrag, (short)0, acc, false, false);
        }
        const float bias = b2[n];
#pragma unroll
        for (int v = 0; v < 8; ++v) {
            int row = v + half * 8;
            float x = acc[v] + bias;
            sH2[row * D2_ + n] = (_Float16)fmaxf(x, 0.0f);
        }
    }
    __syncthreads();

    // --- step 4: s = sigmoid(h2 @ W3 + b3), one row per lane (wave 0)
    if (tid < 16) {
        float acc = b3[0];
        for (int j = 0; j < D2_; ++j)
            acc += (float)sH2[tid * D2_ + j] * W3[j];
        float s = 1.0f / (1.0f + expf(-acc));
        sOut[feat * ROWS_TOTAL + tile * 16 + tid] = s;
    }
}

// ---------------- per-(b,p) head: relu(s@W4[:8]+b4) . W5 -> sigmoid; sqrt(si*st)
// grid = B*P blocks, 64 threads
__global__ void __launch_bounds__(64)
kHead(const float* __restrict__ sOut,   // (2, 32768)
      const float* __restrict__ W4,     // (512, 512) row-major; only rows 0..7 used
      const float* __restrict__ b4,     // (512)
      const float* __restrict__ W5,     // (512)
      const float* __restrict__ b5,     // (1)
      float*       __restrict__ mArr)   // (B*P)
{
    __shared__ float red[64];
    __shared__ float mv[2];
    const int tid = threadIdx.x;
    const int bp  = blockIdx.x;

    for (int f = 0; f < 2; ++f) {
        float sv[8];
#pragma unroll
        for (int l = 0; l < 8; ++l) sv[l] = sOut[f * ROWS_TOTAL + bp * L_ + l];
        float part = 0.0f;
#pragma unroll
        for (int jj = 0; jj < 8; ++jj) {
            int j = tid * 8 + jj;
            float a = b4[j];
#pragma unroll
            for (int l = 0; l < 8; ++l) a += sv[l] * W4[l * D_ + j];
            a = fmaxf(a, 0.0f);
            part += a * W5[j];
        }
        red[tid] = part;
        __syncthreads();
        for (int off = 32; off > 0; off >>= 1) {
            if (tid < off) red[tid] += red[tid + off];
            __syncthreads();
        }
        if (tid == 0) mv[f] = 1.0f / (1.0f + expf(-(red[0] + b5[0])));
        __syncthreads();
    }
    if (tid == 0) mArr[bp] = sqrtf(mv[0] * mv[1]);
}

// ---------------- max over P per batch ----------------
__global__ void __launch_bounds__(64)
kMaxP(const float* __restrict__ mArr, float* __restrict__ out)
{
    __shared__ float red[64];
    const int tid = threadIdx.x;
    red[tid] = mArr[blockIdx.x * P_ + tid];
    __syncthreads();
    for (int off = 32; off > 0; off >>= 1) {
        if (tid < off) red[tid] = fmaxf(red[tid], red[tid + off]);
        __syncthreads();
    }
    if (tid == 0) out[blockIdx.x] = red[0];
}

// ---------------- launch ----------------
extern "C" void kernel_launch(void* const* d_in, const int* in_sizes, int n_in,
                              void* d_out, int out_size, void* d_ws, size_t ws_size,
                              hipStream_t stream) {
    const float* img   = (const float*)d_in[0];
    const float* text  = (const float*)d_in[1];
    const int*   paths = (const int*)  d_in[2];
    const float* W1    = (const float*)d_in[3];
    const float* b1    = (const float*)d_in[4];
    const float* W2    = (const float*)d_in[5];
    const float* b2    = (const float*)d_in[6];
    const float* W3    = (const float*)d_in[7];
    const float* b3    = (const float*)d_in[8];
    const float* W4    = (const float*)d_in[9];
    const float* b4    = (const float*)d_in[10];
    const float* W5    = (const float*)d_in[11];
    const float* b5    = (const float*)d_in[12];
    float* out = (float*)d_out;

    // workspace layout (bytes)
    char* ws = (char*)d_ws;
    const size_t FEAT_H = (size_t)B_ * N_ * D_ * sizeof(_Float16);      // 1 MB
    _Float16* imgH  = (_Float16*)(ws);
    _Float16* textH = (_Float16*)(ws + FEAT_H);
    _Float16* W1t   = (_Float16*)(ws + 2 * FEAT_H);                      // 512x1024 f16 = 1 MB
    _Float16* W2t   = (_Float16*)(ws + 3 * FEAT_H);                      // 256x512  f16 = 256 KB
    float*    sOut  = (float*)   (ws + 3 * FEAT_H + 256 * 1024);        // 2*32768 f32 = 256 KB
    float*    mArr  = (float*)   (ws + 3 * FEAT_H + 512 * 1024);        // 4096 f32

    const int nFeat = B_ * N_ * D_;
    kCvtF16 <<<256, 256, 0, stream>>>(img,  imgH,  nFeat);
    kCvtF16 <<<256, 256, 0, stream>>>(text, textH, nFeat);
    kTransF16<<<256, 256, 0, stream>>>(W1, W1t, K1_, D_);   // (1024,512) -> (512,1024)
    kTransF16<<<256, 256, 0, stream>>>(W2, W2t, D_, D2_);   // (512,256)  -> (256,512)

    kPathMLP<<<2 * TILES, 256, 0, stream>>>(imgH, textH, paths,
                                            W1t, b1, W2t, b2, W3, b3, sOut);

    kHead<<<B_ * P_, 64, 0, stream>>>(sOut, W4, b4, W5, b5, mArr);
    kMaxP<<<B_, 64, 0, stream>>>(mArr, out);
}
